// I2P_MAE_BIND_20993800143424
// MI455X (gfx1250) — matrix operand order repaired and checked by reading, not compiled, not run on hardware
//
#include <hip/hip_runtime.h>
#include <cmath>

// ---------------- constants ----------------
#define BATCH     32
#define NPTS      2048
#define NHEADS    6
#define FPS_MAXN  2048
#define MAXTOK    1572864   // max B*G*D over levels (= 32*512*96 = 32*256*192)

typedef __attribute__((ext_vector_type(16))) _Float16 v16h;
typedef __attribute__((ext_vector_type(8)))  _Float16 v8h;
typedef __attribute__((ext_vector_type(8)))  float    v8f;

__device__ __forceinline__ float gelu_exact(float x) {
    return 0.5f * x * (1.0f + erff(x * 0.70710678118654752f));
}

// ---------------- elementwise add ----------------
__global__ __launch_bounds__(256)
void add_kernel(const float* __restrict__ a, const float* __restrict__ b,
                float* __restrict__ o, int n) {
    int i = blockIdx.x * blockDim.x + threadIdx.x;
    if (i < n) o[i] = a[i] + b[i];
}

// ---------------- weight pack: f32 [K,N] -> f16 WMMA B-fragment layout ----------------
// P[((tN*KB + kblk)*32 + lane)*16 + e] = W[(kblk*32 + (lane>>4)*16 + e)*N + tN*16 + (lane&15)]
__global__ __launch_bounds__(256)
void pack_w_kernel(const float* __restrict__ W, _Float16* __restrict__ P, int Kk, int Nn) {
    int i = blockIdx.x * blockDim.x + threadIdx.x;
    if (i >= Kk * Nn) return;
    int e    = i & 15;
    int lane = (i >> 4) & 31;
    int blk  = i >> 9;               // tN*KB + kblk
    int KB   = Kk >> 5;
    int tN   = blk / KB, kblk = blk % KB;
    int col  = tN * 16 + (lane & 15);
    int kk   = kblk * 32 + (lane >> 4) * 16 + e;
    P[i] = (_Float16)W[(size_t)kk * Nn + col];
}

// ---------------- farthest point sampling ----------------
__global__ __launch_bounds__(256)
void fps_kernel(const float* __restrict__ xyz, int Nn, int S, float* __restrict__ centers) {
    __shared__ float sx[FPS_MAXN], sy[FPS_MAXN], sz[FPS_MAXN], sd[FPS_MAXN];
    __shared__ float rv[256];
    __shared__ int   ri[256];
    int b = blockIdx.x, t = threadIdx.x;
    const float* base = xyz + (size_t)b * Nn * 3;
    for (int i = t; i < Nn; i += 256) {
        sx[i] = base[i * 3 + 0]; sy[i] = base[i * 3 + 1]; sz[i] = base[i * 3 + 2];
        sd[i] = 1e10f;
    }
    __syncthreads();
    int far = 0;
    for (int s = 0; s < S; ++s) {
        float cx = sx[far], cy = sy[far], cz = sz[far];
        if (t == 0) {
            float* c = centers + ((size_t)b * S + s) * 3;
            c[0] = cx; c[1] = cy; c[2] = cz;
        }
        float bv = -1.0f; int bi = 0x7fffffff;
        for (int i = t; i < Nn; i += 256) {
            float dx = sx[i] - cx, dy = sy[i] - cy, dz = sz[i] - cz;
            float d = dx * dx + dy * dy + dz * dz;
            float nd = fminf(sd[i], d); sd[i] = nd;
            if (nd > bv || (nd == bv && i < bi)) { bv = nd; bi = i; }
        }
        rv[t] = bv; ri[t] = bi;
        __syncthreads();
        for (int st = 128; st > 0; st >>= 1) {
            if (t < st) {
                float ov = rv[t + st]; int oi = ri[t + st];
                if (ov > rv[t] || (ov == rv[t] && oi < ri[t])) { rv[t] = ov; ri[t] = oi; }
            }
            __syncthreads();
        }
        far = ri[0];
        __syncthreads();
    }
}

// ---------------- kNN (K smallest d2, ascending, first-index ties) ----------------
__global__ __launch_bounds__(256)
void knn_kernel(const float* __restrict__ src, const float* __restrict__ cen,
                int Nn, int G, int Kc, int* __restrict__ idx) {
    __shared__ float sd[FPS_MAXN];
    __shared__ float rv[256];
    __shared__ int   ri[256];
    int g = blockIdx.x % G, b = blockIdx.x / G, t = threadIdx.x;
    const float* c = cen + ((size_t)b * G + g) * 3;
    float cx = c[0], cy = c[1], cz = c[2];
    const float* base = src + (size_t)b * Nn * 3;
    for (int i = t; i < Nn; i += 256) {
        float dx = base[i * 3 + 0] - cx, dy = base[i * 3 + 1] - cy, dz = base[i * 3 + 2] - cz;
        sd[i] = dx * dx + dy * dy + dz * dz;
    }
    __syncthreads();
    for (int k = 0; k < Kc; ++k) {
        float bv = 1e30f; int bi = 0x7fffffff;
        for (int i = t; i < Nn; i += 256) {
            float v = sd[i];
            if (v < bv || (v == bv && i < bi)) { bv = v; bi = i; }
        }
        rv[t] = bv; ri[t] = bi;
        __syncthreads();
        for (int st = 128; st > 0; st >>= 1) {
            if (t < st) {
                float ov = rv[t + st]; int oi = ri[t + st];
                if (ov < rv[t] || (ov == rv[t] && oi < ri[t])) { rv[t] = ov; ri[t] = oi; }
            }
            __syncthreads();
        }
        if (t == 0) {
            idx[((size_t)b * G + g) * Kc + k] = ri[0];
            sd[ri[0]] = 1e30f;
        }
        __syncthreads();
    }
}

// ---------------- fused mini-PointNet token embed (per group, in LDS) ----------------
extern __shared__ float te_smem[];
__global__ __launch_bounds__(256)
void token_embed_kernel(const float* __restrict__ src, const int* __restrict__ idx,
                        const float* __restrict__ cen, int rel,
                        int Nsrc, int G, int Kc, int Cin, int H, int Dd,
                        const float* __restrict__ c1w, const float* __restrict__ c1b,
                        const float* __restrict__ bn1g, const float* __restrict__ bn1b,
                        const float* __restrict__ c2w, const float* __restrict__ c2b,
                        const float* __restrict__ c3w, const float* __restrict__ c3b,
                        const float* __restrict__ bn2g, const float* __restrict__ bn2b,
                        const float* __restrict__ c4w, const float* __restrict__ c4b,
                        float* __restrict__ tokens) {
    int g = blockIdx.x % G, b = blockIdx.x / G, t = threadIdx.x;
    int buf0n = (Kc * Cin > Kc * H) ? Kc * Cin : Kc * H;
    float* buf0 = te_smem;           // xin, later f2
    float* f1   = buf0 + buf0n;      // K*H
    float* f3   = f1 + Kc * H;       // K*D
    float* gm   = f3 + Kc * Dd;      // H
    const float bnscale = rsqrtf(1.0f + 1e-5f);
    const int* gidx = idx + ((size_t)b * G + g) * Kc;

    for (int e = t; e < Kc * Cin; e += 256) {
        int k = e / Cin, c = e % Cin;
        int r = gidx[k];
        float v = src[((size_t)b * Nsrc + r) * Cin + c];
        if (rel) v -= cen[((size_t)b * G + g) * 3 + c];
        buf0[e] = v;
    }
    __syncthreads();
    for (int e = t; e < Kc * H; e += 256) {     // f1 = relu(bn1(xin@c1w + c1b))
        int k = e / H, j = e % H;
        float acc = c1b[j];
        for (int c = 0; c < Cin; ++c) acc += buf0[k * Cin + c] * c1w[c * H + j];
        acc = acc * (bn1g[j] * bnscale) + bn1b[j];
        f1[e] = acc > 0.0f ? acc : 0.0f;
    }
    __syncthreads();
    for (int e = t; e < Kc * H; e += 256) {     // f2 = f1@c2w + c2b -> buf0
        int k = e / H, j = e % H;
        float acc = c2b[j];
        for (int c = 0; c < H; ++c) acc += f1[k * H + c] * c2w[c * H + j];
        buf0[e] = acc;
    }
    __syncthreads();
    for (int j = t; j < H; j += 256) {          // gmax
        float m = -3.0e38f;
        for (int k = 0; k < Kc; ++k) m = fmaxf(m, buf0[k * H + j]);
        gm[j] = m;
    }
    __syncthreads();
    for (int e = t; e < Kc * Dd; e += 256) {    // f3 = relu(bn2(concat@c3w + c3b))
        int k = e / Dd, j = e % Dd;
        float acc = c3b[j];
        for (int c = 0; c < H; ++c) acc += gm[c] * c3w[c * Dd + j];
        for (int c = 0; c < H; ++c) acc += buf0[k * H + c] * c3w[(H + c) * Dd + j];
        acc = acc * (bn2g[j] * bnscale) + bn2b[j];
        f3[e] = acc > 0.0f ? acc : 0.0f;
    }
    __syncthreads();
    for (int j = t; j < Dd; j += 256) {         // token = max_k (f3@c4w + c4b)
        float m = -3.0e38f;
        for (int k = 0; k < Kc; ++k) {
            float acc = c4b[j];
            for (int c = 0; c < Dd; ++c) acc += f3[k * Dd + c] * c4w[c * Dd + j];
            m = fmaxf(m, acc);
        }
        tokens[((size_t)b * G + g) * Dd + j] = m;
    }
}

// ---------------- pos MLP stage 1 (K=3, gelu, f16 out) ----------------
__global__ __launch_bounds__(256)
void pos1_kernel(const float* __restrict__ cen, const float* __restrict__ w1,
                 const float* __restrict__ b1, _Float16* __restrict__ out, int M, int Dd) {
    int i = blockIdx.x * blockDim.x + threadIdx.x;
    if (i >= M * Dd) return;
    int r = i / Dd, j = i % Dd;
    const float* c = cen + (size_t)r * 3;
    float v = b1[j] + c[0] * w1[0 * Dd + j] + c[1] * w1[1 * Dd + j] + c[2] * w1[2 * Dd + j];
    out[i] = (_Float16)gelu_exact(v);
}

// ---------------- WMMA fp16 GEMM: C[M,N] = act(A[M,K] @ W[K,N] + bias) ----------------
// 2x2 block of 16x16 tiles per wave; A loaded as 2x b128 per frag; B pre-packed
// into per-lane fragment layout (one aligned 32B vector per frag).
// Requires: M, N, K all multiples of 32 (guaranteed by caller).
__global__ __launch_bounds__(256)
void gemm_wmma_f16(const _Float16* __restrict__ A, const _Float16* __restrict__ Wp,
                   const float* __restrict__ bias, float* __restrict__ Cf,
                   _Float16* __restrict__ Ch, int M, int Nn, int Kk, int act) {
    int wave = threadIdx.x >> 5;
    int lane = threadIdx.x & 31;
    int tn2 = Nn >> 5;               // super-tiles along N
    int total = (M >> 5) * tn2;
    int st = blockIdx.x * 8 + wave;
    if (st >= total) return;
    int tM = (st / tn2) * 2;
    int tN = (st % tn2) * 2;
    int low = lane & 15;
    int hi  = lane >> 4;
    int KB  = Kk >> 5;

    const _Float16* Arow0 = A + (size_t)(tM * 16 + low) * Kk;
    const _Float16* Arow1 = Arow0 + (size_t)16 * Kk;
    const v16h* Wb = (const v16h*)Wp;
    const v16h* Wp0 = Wb + ((size_t)tN       * KB) * 32 + lane;
    const v16h* Wp1 = Wb + ((size_t)(tN + 1) * KB) * 32 + lane;

    v8f acc00 = {}, acc01 = {}, acc10 = {}, acc11 = {};
    for (int kb = 0; kb < KB; ++kb) {
        int k0 = kb * 32 + hi * 16;
        v8h a0l = *(const v8h*)(Arow0 + k0);
        v8h a0h = *(const v8h*)(Arow0 + k0 + 8);
        v8h a1l = *(const v8h*)(Arow1 + k0);
        v8h a1h = *(const v8h*)(Arow1 + k0 + 8);
        v16h af0 = __builtin_shufflevector(a0l, a0h, 0,1,2,3,4,5,6,7,8,9,10,11,12,13,14,15);
        v16h af1 = __builtin_shufflevector(a1l, a1h, 0,1,2,3,4,5,6,7,8,9,10,11,12,13,14,15);
        v16h bf0 = Wp0[(size_t)kb * 32];
        v16h bf1 = Wp1[(size_t)kb * 32];
        acc00 = __builtin_amdgcn_wmma_f32_16x16x32_f16(false, af0, false, bf0, (short)0, acc00, false, false);
        acc01 = __builtin_amdgcn_wmma_f32_16x16x32_f16(false, af0, false, bf1, (short)0, acc01, false, false);
        acc10 = __builtin_amdgcn_wmma_f32_16x16x32_f16(false, af1, false, bf0, (short)0, acc10, false, false);
        acc11 = __builtin_amdgcn_wmma_f32_16x16x32_f16(false, af1, false, bf1, (short)0, acc11, false, false);
    }

    auto store_tile = [&](const v8f& acc, int tMi, int tNi) {
#pragma unroll
        for (int i = 0; i < 8; ++i) {
            int m = tMi * 16 + hi * 8 + i;
            int n = tNi * 16 + low;
            float v = acc[i];
            if (bias) v += bias[n];
            if (act == 1) v = gelu_exact(v);
            size_t o = (size_t)m * Nn + n;
            if (Cf) Cf[o] = v;
            if (Ch) Ch[o] = (_Float16)v;
        }
    };
    store_tile(acc00, tM,     tN);
    store_tile(acc01, tM,     tN + 1);
    store_tile(acc10, tM + 1, tN);
    store_tile(acc11, tM + 1, tN + 1);
}

// ---------------- LayerNorm (one wave per row) ----------------
__global__ __launch_bounds__(256)
void ln_kernel(const float* __restrict__ x, const float* __restrict__ g,
               const float* __restrict__ b, float* __restrict__ of,
               _Float16* __restrict__ oh, int M, int Dd) {
    int wave = threadIdx.x >> 5, lane = threadIdx.x & 31;
    int r = blockIdx.x * 8 + wave;
    if (r >= M) return;
    const float* xr = x + (size_t)r * Dd;
    float s = 0.0f, ss = 0.0f;
    for (int j = lane; j < Dd; j += 32) { float v = xr[j]; s += v; ss += v * v; }
    for (int o = 16; o > 0; o >>= 1) { s += __shfl_xor(s, o, 32); ss += __shfl_xor(ss, o, 32); }
    float mean = s / (float)Dd;
    float var = ss / (float)Dd - mean * mean;
    if (var < 0.0f) var = 0.0f;
    float rstd = rsqrtf(var + 1e-5f);
    for (int j = lane; j < Dd; j += 32) {
        float v = (xr[j] - mean) * rstd * g[j] + b[j];
        if (of) of[(size_t)r * Dd + j] = v;
        if (oh) oh[(size_t)r * Dd + j] = (_Float16)v;
    }
}

// ---------------- fused masked flash attention (per batch,head) ----------------
__global__ __launch_bounds__(256)
void attn_kernel(const float* __restrict__ qkv, const float* __restrict__ cen,
                 _Float16* __restrict__ out, int G, int hd, int Dd,
                 float scale, float rad2) {
    __shared__ float sK[8192], sV[8192], sC[1536];
    int b = blockIdx.x / NHEADS, h = blockIdx.x % NHEADS;
    int t = threadIdx.x, lane = t & 31, wave = t >> 5;
    int D3 = 3 * Dd;
    for (int e = t; e < G * hd; e += 256) {
        int m = e / hd, c = e % hd;
        sK[e] = qkv[((size_t)b * G + m) * D3 +     Dd + h * hd + c];
        sV[e] = qkv[((size_t)b * G + m) * D3 + 2 * Dd + h * hd + c];
    }
    for (int e = t; e < G * 3; e += 256) sC[e] = cen[(size_t)b * G * 3 + e];
    __syncthreads();
    int cnt = G >> 5;
    float sc[16];
    for (int n = wave; n < G; n += 8) {
        float q0 = (lane < hd)      ? qkv[((size_t)b * G + n) * D3 + h * hd + lane]      : 0.0f;
        float q1 = (lane + 32 < hd) ? qkv[((size_t)b * G + n) * D3 + h * hd + lane + 32] : 0.0f;
        float nx = sC[n * 3], ny = sC[n * 3 + 1], nz = sC[n * 3 + 2];
        float rmax = -3.0e38f;
        for (int i = 0; i < cnt; ++i) {
            int m = i * 32 + lane;
            float dot = 0.0f;
            for (int c = 0; c < hd; ++c) {
                float qv = (c < 32) ? __shfl(q0, c, 32) : __shfl(q1, c - 32, 32);
                dot += qv * sK[m * hd + c];
            }
            dot *= scale;
            float dx = sC[m * 3] - nx, dy = sC[m * 3 + 1] - ny, dz = sC[m * 3 + 2] - nz;
            if (dx * dx + dy * dy + dz * dz >= rad2) dot = -3.0e38f;
            sc[i] = dot;
            rmax = fmaxf(rmax, dot);
        }
        for (int o = 16; o > 0; o >>= 1) rmax = fmaxf(rmax, __shfl_xor(rmax, o, 32));
        float sum = 0.0f;
        for (int i = 0; i < cnt; ++i) { float e = expf(sc[i] - rmax); sc[i] = e; sum += e; }
        for (int o = 16; o > 0; o >>= 1) sum += __shfl_xor(sum, o, 32);
        float inv = 1.0f / sum;
        float a0 = 0.0f, a1 = 0.0f;
        for (int m = 0; m < G; ++m) {
            float p = __shfl(sc[m >> 5], m & 31, 32);
            if (lane < hd)      a0 += p * sV[m * hd + lane];
            if (lane + 32 < hd) a1 += p * sV[m * hd + lane + 32];
        }
        if (lane < hd)      out[((size_t)b * G + n) * Dd + h * hd + lane]      = (_Float16)(a0 * inv);
        if (lane + 32 < hd) out[((size_t)b * G + n) * Dd + h * hd + lane + 32] = (_Float16)(a1 * inv);
    }
}

// ---------------- final mean + max pooling over tokens ----------------
__global__ __launch_bounds__(256)
void pool_kernel(const float* __restrict__ x, float* __restrict__ out,
                 int G, int Dd, int Btot) {
    int i = blockIdx.x * blockDim.x + threadIdx.x;
    if (i >= Btot * Dd) return;
    int b = i / Dd, j = i % Dd;
    float s = 0.0f, m = -3.0e38f;
    for (int n = 0; n < G; ++n) {
        float v = x[((size_t)b * G + n) * Dd + j];
        s += v; m = fmaxf(m, v);
    }
    out[i] = s / (float)G + m;
}

// =====================================================================
//                              host side
// =====================================================================
struct BlkP {
    const float *fc1_b, *fc1_w, *fc2_b, *fc2_w, *ln1_b, *ln1_g,
                *ln2_b, *ln2_g, *proj_b, *proj_w, *qkv_w;
};
struct LvlP {
    BlkP blk[5];
    const float *pos_b1, *pos_b2, *pos_w1, *pos_w2;
    const float *bn1_b, *bn1_g, *bn2_b, *bn2_g,
                *c1_b, *c1_w, *c2_b, *c2_w, *c3_b, *c3_w, *c4_b, *c4_w;
};

extern "C" void kernel_launch(void* const* d_in, const int* in_sizes, int n_in,
                              void* d_out, int out_size, void* d_ws, size_t ws_size,
                              hipStream_t stream) {
    (void)in_sizes; (void)out_size; (void)ws_size;
    if (n_in < 216) return;

    static const int   Gs[3] = {512, 256, 64};
    static const int   Ks[3] = {48, 24, 24};
    static const int   Dsz[3] = {96, 192, 384};
    static const float RADII[3] = {0.24f, 0.48f, 0.96f};

    // ---- parse inputs (JAX dict flatten: sorted keys) ----
    const float* pts = (const float*)d_in[0];
    int p = 1;
    LvlP L[3];
    for (int i = 0; i < 3; ++i) {
        for (int j = 0; j < 5; ++j) {
            BlkP& k = L[i].blk[j];
            k.fc1_b  = (const float*)d_in[p++]; k.fc1_w  = (const float*)d_in[p++];
            k.fc2_b  = (const float*)d_in[p++]; k.fc2_w  = (const float*)d_in[p++];
            k.ln1_b  = (const float*)d_in[p++]; k.ln1_g  = (const float*)d_in[p++];
            k.ln2_b  = (const float*)d_in[p++]; k.ln2_g  = (const float*)d_in[p++];
            k.proj_b = (const float*)d_in[p++]; k.proj_w = (const float*)d_in[p++];
            k.qkv_w  = (const float*)d_in[p++];
        }
        L[i].pos_b1 = (const float*)d_in[p++]; L[i].pos_b2 = (const float*)d_in[p++];
        L[i].pos_w1 = (const float*)d_in[p++]; L[i].pos_w2 = (const float*)d_in[p++];
        L[i].bn1_b = (const float*)d_in[p++]; L[i].bn1_g = (const float*)d_in[p++];
        L[i].bn2_b = (const float*)d_in[p++]; L[i].bn2_g = (const float*)d_in[p++];
        L[i].c1_b  = (const float*)d_in[p++]; L[i].c1_w  = (const float*)d_in[p++];
        L[i].c2_b  = (const float*)d_in[p++]; L[i].c2_w  = (const float*)d_in[p++];
        L[i].c3_b  = (const float*)d_in[p++]; L[i].c3_w  = (const float*)d_in[p++];
        L[i].c4_b  = (const float*)d_in[p++]; L[i].c4_w  = (const float*)d_in[p++];
    }
    const float* norm_b = (const float*)d_in[p++];
    const float* norm_g = (const float*)d_in[p++];

    // ---- workspace bump allocator ----
    char* wsb = (char*)d_ws;
    size_t off = 0;
    auto alloc = [&](size_t bytes) -> void* {
        void* q = wsb + off;
        off += ((bytes + 255) / 256) * 256;
        return q;
    };

    float* centers[3];
    for (int i = 0; i < 3; ++i) centers[i] = (float*)alloc((size_t)BATCH * Gs[i] * 3 * 4);
    int*       knnIdx = (int*)alloc((size_t)BATCH * 512 * 48 * 4);
    float*     xb0    = (float*)alloc((size_t)MAXTOK * 4);
    float*     xb1    = (float*)alloc((size_t)MAXTOK * 4);
    float*     tmpF   = (float*)alloc((size_t)MAXTOK * 4);
    float*     posb   = (float*)alloc((size_t)MAXTOK * 4);
    float*     qkvb   = (float*)alloc((size_t)MAXTOK * 3 * 4);
    _Float16*  h16a   = (_Float16*)alloc((size_t)MAXTOK * 2);
    _Float16*  h16b   = (_Float16*)alloc((size_t)MAXTOK * 2);
    _Float16*  h16fc  = (_Float16*)alloc((size_t)MAXTOK * 4 * 2);

    auto pack = [&](const float* src, _Float16* dst, int Kk, int Nn) {
        int n = Kk * Nn;
        pack_w_kernel<<<(n + 255) / 256, 256, 0, stream>>>(src, dst, Kk, Nn);
    };

    // fp16 packed weight copies (WMMA B-fragment layout)
    _Float16 *qkvW[3][5], *projW[3][5], *fc1W[3][5], *fc2W[3][5], *posW2[3];
    for (int i = 0; i < 3; ++i) {
        int d = Dsz[i];
        for (int j = 0; j < 5; ++j) {
            qkvW[i][j]  = (_Float16*)alloc((size_t)d * 3 * d * 2);
            projW[i][j] = (_Float16*)alloc((size_t)d * d * 2);
            fc1W[i][j]  = (_Float16*)alloc((size_t)d * 4 * d * 2);
            fc2W[i][j]  = (_Float16*)alloc((size_t)4 * d * d * 2);
            pack(L[i].blk[j].qkv_w,  qkvW[i][j],  d, 3 * d);
            pack(L[i].blk[j].proj_w, projW[i][j], d, d);
            pack(L[i].blk[j].fc1_w,  fc1W[i][j],  d, 4 * d);
            pack(L[i].blk[j].fc2_w,  fc2W[i][j],  4 * d, d);
        }
        posW2[i] = (_Float16*)alloc((size_t)d * d * 2);
        pack(L[i].pos_w2, posW2[i], d, d);
    }

    auto gemm = [&](const _Float16* A, const _Float16* W, const float* bias,
                    float* Cf, _Float16* Ch, int M, int Nn, int Kk, int act) {
        int tiles = (M / 32) * (Nn / 32);        // 2x2 super-tiles per wave
        gemm_wmma_f16<<<(tiles + 7) / 8, 256, 0, stream>>>(A, W, bias, Cf, Ch, M, Nn, Kk, act);
    };

    // ---- hierarchical pipeline ----
    const float* srcXYZ = pts;
    int Nsrc = NPTS;
    float* xv = xb0;   // current features
    float* xh = xb1;   // temp (h)

    for (int i = 0; i < 3; ++i) {
        int G = Gs[i], K = Ks[i], D = Dsz[i], H = D / 2, M = BATCH * G;

        fps_kernel<<<BATCH, 256, 0, stream>>>(srcXYZ, Nsrc, G, centers[i]);
        knn_kernel<<<BATCH * G, 256, 0, stream>>>(srcXYZ, centers[i], Nsrc, G, K, knnIdx);

        int Cin = (i == 0) ? 3 : Dsz[i - 1];
        const float* featSrc = (i == 0) ? pts : xv;
        int featN = (i == 0) ? NPTS : Gs[i - 1];
        float* tokDst = (i == 0) ? xv : xh;
        {
            int buf0n = (K * Cin > K * H) ? K * Cin : K * H;
            size_t smemB = (size_t)(buf0n + K * H + K * D + H) * 4;
            token_embed_kernel<<<BATCH * G, 256, smemB, stream>>>(
                featSrc, knnIdx, centers[i], (i == 0) ? 1 : 0,
                featN, G, K, Cin, H, D,
                L[i].c1_w, L[i].c1_b, L[i].bn1_g, L[i].bn1_b,
                L[i].c2_w, L[i].c2_b,
                L[i].c3_w, L[i].c3_b, L[i].bn2_g, L[i].bn2_b,
                L[i].c4_w, L[i].c4_b, tokDst);
        }
        if (i > 0) { float* t = xv; xv = tokDst; xh = t; }

        // positional embedding
        pos1_kernel<<<(M * D + 255) / 256, 256, 0, stream>>>(centers[i], L[i].pos_w1,
                                                             L[i].pos_b1, h16a, M, D);
        gemm(h16a, posW2[i], L[i].pos_b2, posb, nullptr, M, D, D, 0);

        // encoder blocks
        int hd = D / NHEADS;
        float scale = 1.0f / sqrtf((float)hd);
        float rad2 = RADII[i] * RADII[i];
        for (int j = 0; j < 5; ++j) {
            const BlkP& bp = L[i].blk[j];
            add_kernel<<<(M * D + 255) / 256, 256, 0, stream>>>(xv, posb, xh, M * D);
            ln_kernel<<<(M + 7) / 8, 256, 0, stream>>>(xh, bp.ln1_g, bp.ln1_b,
                                                       nullptr, h16a, M, D);
            gemm(h16a, qkvW[i][j], nullptr, qkvb, nullptr, M, 3 * D, D, 0);
            attn_kernel<<<BATCH * NHEADS, 256, 0, stream>>>(qkvb, centers[i], h16b,
                                                            G, hd, D, scale, rad2);
            gemm(h16b, projW[i][j], bp.proj_b, tmpF, nullptr, M, D, D, 0);
            add_kernel<<<(M * D + 255) / 256, 256, 0, stream>>>(xh, tmpF, xv, M * D);
            ln_kernel<<<(M + 7) / 8, 256, 0, stream>>>(xv, bp.ln2_g, bp.ln2_b,
                                                       nullptr, h16a, M, D);
            gemm(h16a, fc1W[i][j], bp.fc1_b, nullptr, h16fc, M, 4 * D, D, 1);
            gemm(h16fc, fc2W[i][j], bp.fc2_b, tmpF, nullptr, M, D, 4 * D, 0);
            add_kernel<<<(M * D + 255) / 256, 256, 0, stream>>>(xv, tmpF, xv, M * D);
        }

        srcXYZ = centers[i];
        Nsrc = G;
    }

    // final norm + mean/max pooling
    {
        int M = BATCH * Gs[2], D = Dsz[2];
        ln_kernel<<<(M + 7) / 8, 256, 0, stream>>>(xv, norm_g, norm_b, tmpF, nullptr, M, D);
        pool_kernel<<<(BATCH * D + 255) / 256, 256, 0, stream>>>(tmpF, (float*)d_out,
                                                                 Gs[2], D, BATCH);
    }
}
